// SparseAutoencoder_91336774516825
// MI455X (gfx1250) — compile-verified, matrix operand order checked by reading, not run on hardware
//
#include <hip/hip_runtime.h>
#include <hip/hip_bf16.h>

// Problem constants (reference: B=4096, D=768, F=24576, k=32)
#define BATCH  4096
#define DMODEL 768
#define NFEAT  24576
#define TOPK   32

typedef __attribute__((ext_vector_type(16))) __bf16 v16bf;
typedef __attribute__((ext_vector_type(8)))  __bf16 v8bf;
typedef __attribute__((ext_vector_type(8)))  float  v8f;

// LDS row stride in bf16 elements: 40 halves = 80 bytes = 20 banks.
// 20*r mod 64 for r=0..15 hits 16 disjoint 4-bank b128 spans -> conflict-free.
#define LDS_STRIDE 40

// Build a 16x32 bf16 WMMA A/B fragment from an LDS row (N- or M-major tile).
// Hardware layout (ISA 7.12.2, 16-bit A 16x32): lanes 0-15 hold K{0..7,16..23},
// lanes 16-31 hold K{8..15,24..31} -> two contiguous 16B chunks per lane.
__device__ __forceinline__ v16bf frag_from_lds(const __bf16* rowbase, int half) {
  const v8bf lo = *(const v8bf*)(rowbase + half * 8);
  const v8bf hi = *(const v8bf*)(rowbase + 16 + half * 8);
  return __builtin_shufflevector(lo, hi, 0, 1, 2, 3, 4, 5, 6, 7,
                                 8, 9, 10, 11, 12, 13, 14, 15);
}

// ---------------------------------------------------------------------------
// Kernel 1: pre = ReLU((x - b_pre) @ W_enc + b_enc), written dense into h.
// 128x128 tile / workgroup, 8 waves (2x4), each wave: 4x2 WMMA 16x16 tiles.
// bf16 operands (converted during staging), f32 accumulate via
// v_wmma_f32_16x16x32_bf16. Register-software-pipelined + double-buffered LDS:
// next K-slab's global loads are issued BEFORE the WMMA block (no consumer ->
// they stay in flight across the matrix math), then waited/converted/stored
// to the spare LDS buffer after compute.
// ---------------------------------------------------------------------------
__global__ __launch_bounds__(256) void sae_enc_gemm(
    const float* __restrict__ x, const float* __restrict__ b_pre,
    const float* __restrict__ W_enc, const float* __restrict__ b_enc,
    float* __restrict__ h)
{
  __shared__ __bf16 sA[2][128 * LDS_STRIDE];  // 20 KB
  __shared__ __bf16 sB[2][128 * LDS_STRIDE];  // 20 KB

  const int tid    = threadIdx.x;
  const int blockM = blockIdx.y * 128;
  const int blockN = blockIdx.x * 128;
  const int wid    = tid >> 5;     // 8 waves (wave32)
  const int lane   = tid & 31;
  const int half   = lane >> 4;    // 0 / 1
  const int l      = lane & 15;
  const int m_base = (wid >> 2) * 64;  // wave M offset (0 / 64)
  const int n_base = (wid & 3) * 32;   // wave N offset (0/32/64/96)

  v8f acc[4][2] = {};

  // Each thread owns 4 A-slots (r = s>>3, kq = s&7) and 4 B-slots
  // (kr = s>>5, nq = s&31), s = tid + 256*i. float4-coalesced global loads.
  float4 aReg[4], bReg[4];

  auto issue_loads = [&](int k0) {
    #pragma unroll
    for (int i = 0; i < 4; ++i) {
      const int s = tid + 256 * i;
      const int r = s >> 3, q = s & 7;
      aReg[i] = *(const float4*)(x + (size_t)(blockM + r) * DMODEL + k0 + q * 4);
      const int kr = s >> 5, nq = s & 31;
      bReg[i] = *(const float4*)(W_enc + (size_t)(k0 + kr) * NFEAT + blockN + nq * 4);
    }
  };

  auto convert_store = [&](int buf, int k0) {
    #pragma unroll
    for (int i = 0; i < 4; ++i) {
      const int s = tid + 256 * i;
      // A: fused (x - b_pre), bf16 convert, M-major LDS tile
      const int r = s >> 3, q = s & 7;
      const float4 bp = *(const float4*)(b_pre + k0 + q * 4);  // L0-resident
      __bf16* d = &sA[buf][r * LDS_STRIDE + q * 4];
      d[0] = (__bf16)(aReg[i].x - bp.x); d[1] = (__bf16)(aReg[i].y - bp.y);
      d[2] = (__bf16)(aReg[i].z - bp.z); d[3] = (__bf16)(aReg[i].w - bp.w);
      // B: transpose to N-major LDS tile (contiguous K per N row)
      const int kr = s >> 5, nq = s & 31;
      sB[buf][(nq * 4 + 0) * LDS_STRIDE + kr] = (__bf16)bReg[i].x;
      sB[buf][(nq * 4 + 1) * LDS_STRIDE + kr] = (__bf16)bReg[i].y;
      sB[buf][(nq * 4 + 2) * LDS_STRIDE + kr] = (__bf16)bReg[i].z;
      sB[buf][(nq * 4 + 3) * LDS_STRIDE + kr] = (__bf16)bReg[i].w;
    }
  };

  issue_loads(0);
  convert_store(0, 0);
  __syncthreads();

  const int KSTEPS = DMODEL / 32;  // 24
  for (int kt = 0; kt < KSTEPS; ++kt) {
    const int cur = kt & 1;
    const bool have_next = (kt + 1 < KSTEPS);
    if (have_next) issue_loads((kt + 1) * 32);  // in flight across WMMAs

    v16bf afrag[4], bfrag[2];
    #pragma unroll
    for (int mt = 0; mt < 4; ++mt)
      afrag[mt] = frag_from_lds(&sA[cur][(m_base + mt * 16 + l) * LDS_STRIDE], half);
    #pragma unroll
    for (int nt = 0; nt < 2; ++nt)
      bfrag[nt] = frag_from_lds(&sB[cur][(n_base + nt * 16 + l) * LDS_STRIDE], half);

    #pragma unroll
    for (int mt = 0; mt < 4; ++mt)
      #pragma unroll
      for (int nt = 0; nt < 2; ++nt)
        acc[mt][nt] = __builtin_amdgcn_wmma_f32_16x16x32_bf16(
            false, afrag[mt], false, bfrag[nt], (short)0, acc[mt][nt],
            false, false);

    if (have_next) convert_store(cur ^ 1, (kt + 1) * 32);
    __syncthreads();
  }

  // Epilogue: + b_enc, ReLU, store f32.
  // C layout (ISA 7.12.2): VGPR r, lanes 0-15 -> M = r, lanes 16-31 -> M = r+8.
  #pragma unroll
  for (int nt = 0; nt < 2; ++nt) {
    const int gcol = blockN + n_base + nt * 16 + l;
    const float bias = b_enc[gcol];
    #pragma unroll
    for (int mt = 0; mt < 4; ++mt) {
      #pragma unroll
      for (int r = 0; r < 8; ++r) {
        const int grow = blockM + m_base + mt * 16 + half * 8 + r;
        const float v = acc[mt][nt][r] + bias;
        h[(size_t)grow * NFEAT + gcol] = v > 0.f ? v : 0.f;
      }
    }
  }
}

// ---------------------------------------------------------------------------
// Kernel 2: per-row top-32 over F=24576 with the row resident in LDS (96 KB,
// fine inside CDNA5's 320 KB/WGP). 32 rounds of block argmax (tie -> lowest
// index, matching lax.top_k), then rewrite the row sparse in place.
// ---------------------------------------------------------------------------
__global__ __launch_bounds__(256) void sae_topk(
    float* __restrict__ h, float* __restrict__ vals, int* __restrict__ idxs)
{
  extern __shared__ unsigned char smem[];
  float* sRow  = (float*)smem;           // NFEAT
  float* red_v = sRow + NFEAT;           // 256
  int*   red_i = (int*)(red_v + 256);    // 256
  float* sel_v = (float*)(red_i + 256);  // TOPK
  int*   sel_i = (int*)(sel_v + TOPK);   // TOPK

  const int tid = threadIdx.x;
  const size_t rowoff = (size_t)blockIdx.x * NFEAT;

  for (int i = tid; i < NFEAT; i += 256) sRow[i] = h[rowoff + i];
  __syncthreads();

  for (int it = 0; it < TOPK; ++it) {
    float bv = -1.f; int bi = NFEAT;  // values are post-ReLU (>= 0)
    for (int i = tid; i < NFEAT; i += 256) {
      const float v = sRow[i];
      if (v > bv) { bv = v; bi = i; }  // strict '>' keeps lowest index on ties
    }
    red_v[tid] = bv; red_i[tid] = bi;
    __syncthreads();
    #pragma unroll
    for (int s = 128; s > 0; s >>= 1) {
      if (tid < s) {
        const float ov = red_v[tid + s]; const int oi = red_i[tid + s];
        if (ov > red_v[tid] || (ov == red_v[tid] && oi < red_i[tid])) {
          red_v[tid] = ov; red_i[tid] = oi;
        }
      }
      __syncthreads();
    }
    if (tid == 0) {
      sel_v[it] = red_v[0]; sel_i[it] = red_i[0];
      sRow[red_i[0]] = -2.f;  // remove winner; safe sentinel below ReLU range
    }
    __syncthreads();
  }

  if (tid < TOPK) {
    vals[(size_t)blockIdx.x * TOPK + tid] = sel_v[tid];
    idxs[(size_t)blockIdx.x * TOPK + tid] = sel_i[tid];
  }

  float lv[TOPK]; int li[TOPK];
  #pragma unroll
  for (int j = 0; j < TOPK; ++j) { lv[j] = sel_v[j]; li[j] = sel_i[j]; }

  for (int i = tid; i < NFEAT; i += 256) {
    float o = 0.f;
    #pragma unroll
    for (int j = 0; j < TOPK; ++j) o = (i == li[j]) ? lv[j] : o;
    h[rowoff + i] = o;
  }
}

// ---------------------------------------------------------------------------
// Kernel 3: x_hat[row] = sum_j sel_v[j] * W_dec[sel_i[j], :] + b_dec.
// One block per row; W_dec (75.5 MB) is L2-resident on the 192 MB L2.
// ---------------------------------------------------------------------------
__global__ __launch_bounds__(256) void sae_decode(
    const float* __restrict__ vals, const int* __restrict__ idxs,
    const float* __restrict__ W_dec, const float* __restrict__ b_dec,
    float* __restrict__ xhat)
{
  __shared__ float sv[TOPK];
  __shared__ int   si[TOPK];
  const int tid = threadIdx.x;
  const int row = blockIdx.x;
  if (tid < TOPK) {
    sv[tid] = vals[(size_t)row * TOPK + tid];
    si[tid] = idxs[(size_t)row * TOPK + tid];
  }
  __syncthreads();

  float a0 = b_dec[tid];
  float a1 = b_dec[tid + 256];
  float a2 = b_dec[tid + 512];
  #pragma unroll 8
  for (int j = 0; j < TOPK; ++j) {
    const float w = sv[j];
    const float* wr = W_dec + (size_t)si[j] * DMODEL;
    a0 = fmaf(w, wr[tid], a0);
    a1 = fmaf(w, wr[tid + 256], a1);
    a2 = fmaf(w, wr[tid + 512], a2);
  }
  xhat[(size_t)row * DMODEL + tid]       = a0;
  xhat[(size_t)row * DMODEL + tid + 256] = a1;
  xhat[(size_t)row * DMODEL + tid + 512] = a2;
}

extern "C" void kernel_launch(void* const* d_in, const int* in_sizes, int n_in,
                              void* d_out, int out_size, void* d_ws, size_t ws_size,
                              hipStream_t stream) {
  (void)in_sizes; (void)n_in; (void)out_size; (void)ws_size;
  const float* x     = (const float*)d_in[0];
  const float* b_pre = (const float*)d_in[1];
  const float* W_enc = (const float*)d_in[2];
  const float* b_enc = (const float*)d_in[3];
  const float* W_dec = (const float*)d_in[4];
  const float* b_dec = (const float*)d_in[5];
  // d_in[6] is k == 32, baked into TOPK.

  float* h    = (float*)d_out;                 // [B, F] (written fully)
  float* xhat = h + (size_t)BATCH * NFEAT;     // [B, D]

  float* vals = (float*)d_ws;                                        // B*32 f32
  int*   idxs = (int*)((char*)d_ws + (size_t)BATCH * TOPK * sizeof(float));

  dim3 ggrid(NFEAT / 128, BATCH / 128);  // (192, 32)
  sae_enc_gemm<<<ggrid, 256, 0, stream>>>(x, b_pre, W_enc, b_enc, h);

  const size_t topk_smem =
      (size_t)NFEAT * sizeof(float) + 256 * sizeof(float) + 256 * sizeof(int) +
      TOPK * sizeof(float) + TOPK * sizeof(int);
  sae_topk<<<BATCH, 256, topk_smem, stream>>>(h, vals, idxs);

  sae_decode<<<BATCH, 256, 0, stream>>>(vals, idxs, W_dec, b_dec, xhat);
}
